// DCN_Align_47820165874339
// MI455X (gfx1250) — matrix operand order repaired and checked by reading, not compiled, not run on hardware
//
#include <hip/hip_runtime.h>
#include <math.h>

typedef __attribute__((ext_vector_type(16))) _Float16 v16h;
typedef __attribute__((ext_vector_type(8)))  float    v8f;

static __device__ __forceinline__ int imin(int a, int b) { return a < b ? a : b; }
static __device__ __forceinline__ int imax(int a, int b) { return a > b ? a : b; }

// ---------------------------------------------------------------------------
// Pack fp32 OIHW weights into WMMA A-fragment layout (f16), zero-padded.
// Layout: [mchunk][kchunk][lane 0..31][elem 0..15] contiguous halves.
// A 16x32 f16 layout (contiguous k): lane<16: K = e; lane>=16: K = 16 + e.
// M = mchunk*16 + (lane&15).
// ---------------------------------------------------------------------------
__global__ __launch_bounds__(256) void pack_w_kernel(const float* __restrict__ w,
                                                     _Float16* __restrict__ out,
                                                     int Cout, int Kin) {
  int nK = (Kin + 31) >> 5;
  int nM = (Cout + 15) >> 4;
  long total = (long)nM * nK * 512;
  for (long i = (long)blockIdx.x * 256 + threadIdx.x; i < total;
       i += (long)gridDim.x * 256) {
    int e = (int)(i & 15);
    int lane = (int)((i >> 4) & 31);
    long t = i >> 9;
    int kc = (int)(t % nK);
    int mc = (int)(t / nK);
    int M = mc * 16 + (lane & 15);
    int K = kc * 32 + e + ((lane >> 4) << 4);
    float v = 0.0f;
    if (M < Cout && K < Kin) v = w[(long)M * Kin + K];
    out[i] = (_Float16)v;
  }
}

// ---------------------------------------------------------------------------
// Implicit-GEMM conv via WMMA f32<-f16, fully constant-folded addressing.
//   GEMM: M = COUT, N = 4*HOUT*WOUT, K = CIN*KS*KS
//   Block = 128 threads = 4 waves; wave w -> M-chunk blockIdx.y*4+w.
//   Each block covers a 16*NT-pixel N tile; each wave issues NT WMMAs per
//   K-chunk reusing one A fragment (weights).
//   Channel-concat: c < C0 reads in0, else in1.  act: 0 none, 1 lrelu, 2 sigmoid.
// ---------------------------------------------------------------------------
template <int KS, int STRIDE, int C0, int CIN, int COUT, int HIN, int WIN, int NT>
__global__ __launch_bounds__(128) void conv_wmma_kernel(
    const float* __restrict__ in0, const float* __restrict__ in1,
    const _Float16* __restrict__ wpack, const float* __restrict__ bias,
    float* __restrict__ out, int act) {
  constexpr int PAD  = (KS - 1) / 2;
  constexpr int HOUT = (HIN + 2 * PAD - KS) / STRIDE + 1;
  constexpr int WOUT = (WIN + 2 * PAD - KS) / STRIDE + 1;
  constexpr int KSQ  = KS * KS;
  constexpr int KIN  = CIN * KSQ;
  constexpr int NK   = KIN / 32;
  constexpr int NM   = (COUT + 15) / 16;
  constexpr int HW   = HOUT * WOUT;
  constexpr int C1   = CIN - C0;
  static_assert(KIN % 32 == 0, "K must tile by 32");

  const int lane   = threadIdx.x & 31;
  const int wv     = threadIdx.x >> 5;
  const int mchunk = blockIdx.y * 4 + wv;
  const int tile0  = blockIdx.x * (16 * NT);

  __shared__ _Float16 ldsB[16 * NT * 32];   // [n][k]: B-frag = 16 contiguous halves

  v8f acc[NT];
#pragma unroll
  for (int t = 0; t < NT; ++t) acc[t] = {};

  for (int kc = 0; kc < NK; ++kc) {
    __syncthreads();
    // cooperative im2col fill of 32(K) x 16*NT(N), fp32 -> f16
#pragma unroll 4
    for (int i = threadIdx.x; i < 512 * NT; i += 128) {
      int nn = i >> 5;
      int kk = i & 31;
      int K  = kc * 32 + kk;
      int np = tile0 + nn;
      int b  = np / HW;          // HW is a power of two -> shift
      int rp = np % HW;
      int oy = rp / WOUT;        // WOUT power of two -> shift
      int ox = rp % WOUT;
      int c  = K / KSQ;          // KSQ in {1,9} -> folded / magic-mul
      int tt = K % KSQ;
      int r  = tt / KS;
      int s  = tt % KS;
      int iy = oy * STRIDE + r - PAD;
      int ix = ox * STRIDE + s - PAD;
      float v = 0.0f;
      if (iy >= 0 && iy < HIN && ix >= 0 && ix < WIN) {
        const float* src = (c < C0) ? in0 : in1;
        int cc   = (c < C0) ? c  : c - C0;
        int Csrc = (c < C0) ? C0 : C1;
        v = src[(((size_t)b * Csrc + cc) * HIN + iy) * WIN + ix];
      }
      ldsB[i] = (_Float16)v;
    }
    __syncthreads();
    v16h afrag = {};
    if (mchunk < NM) {
      afrag = *(const v16h*)(wpack + (((size_t)mchunk * NK + kc) * 32 + lane) * 16);
    }
#pragma unroll
    for (int t = 0; t < NT; ++t) {
      const v16h* bp =
          (const v16h*)&ldsB[((lane & 15) + 16 * t) * 32 + ((lane >> 4) << 4)];
      acc[t] = __builtin_amdgcn_wmma_f32_16x16x32_f16(false, afrag, false, *bp,
                                                      (short)0, acc[t], false, false);
    }
  }
  if (mchunk >= NM) return;
#pragma unroll
  for (int t = 0; t < NT; ++t) {
    int np = tile0 + t * 16 + (lane & 15);
    int b  = np / HW;
    int rp = np % HW;
    int oy = rp / WOUT;
    int ox = rp % WOUT;
#pragma unroll
    for (int v = 0; v < 8; ++v) {
      int M = mchunk * 16 + v + ((lane >> 4) << 3);
      if (M < COUT) {
        float xv = acc[t][v] + bias[M];
        if (act == 1)      xv = xv >= 0.0f ? xv : 0.1f * xv;
        else if (act == 2) xv = 1.0f / (1.0f + __expf(-xv));
        out[(((size_t)b * COUT + M) * HOUT + oy) * WOUT + ox] = xv;
      }
    }
  }
}

template <int KS, int STRIDE, int C0, int CIN, int COUT, int HIN, int WIN>
static void launch_conv(const float* in0, const float* in1, const _Float16* wp,
                        const float* bias, float* out, int act, hipStream_t s) {
  constexpr int PAD  = (KS - 1) / 2;
  constexpr int HOUT = (HIN + 2 * PAD - KS) / STRIDE + 1;
  constexpr int WOUT = (WIN + 2 * PAD - KS) / STRIDE + 1;
  constexpr int NT   = 4;
  constexpr int NM   = (COUT + 15) / 16;
  int ntiles = (4 * HOUT * WOUT) / (16 * NT);
  int mb     = (NM + 3) / 4;
  conv_wmma_kernel<KS, STRIDE, C0, CIN, COUT, HIN, WIN, NT>
      <<<dim3(ntiles, mb), 128, 0, s>>>(in0, in1, wp, bias, out, act);
}

// ---------------------------------------------------------------------------
// Modulated deformable conv 5x5, DG=4, H=W=64, B=4, via WMMA.
//   Phase 1: bilinear-sample + modulate a [N=16][K=1600] f16 tile into LDS.
//   Phase 2: GEMM M=64 (4 waves x 16), K=1600 (50 chunks), N=16.
// ---------------------------------------------------------------------------
__global__ __launch_bounds__(128) void dcn_wmma_kernel(
    const float* __restrict__ x, const float* __restrict__ offx,
    const float* __restrict__ offy, const float* __restrict__ msk,
    const _Float16* __restrict__ wpack, const float* __restrict__ bias,
    float* __restrict__ out) {
  constexpr int Hh = 64, Ww = 64, hw = 4096;
  const int lane  = threadIdx.x & 31;
  const int wv    = threadIdx.x >> 5;
  const int tile0 = blockIdx.x * 16;

  __shared__ _Float16 samp[16][1600];   // [n][K = c*25 + k]

  for (int task = threadIdx.x; task < 1600; task += 128) {   // (dg,k,n)
    int dg = task / 400;
    int rm = task - dg * 400;
    int k  = rm >> 4;
    int n  = rm & 15;
    int np = tile0 + n;
    int b  = np / hw;
    int rp = np % hw;
    int yy = rp / Ww;
    int xx = rp % Ww;
    size_t obase = (((size_t)b * 100 + dg * 25 + k) * Hh + yy) * Ww + xx;
    float ofx = offx[obase];
    float ofy = offy[obase];
    float mm  = msk[obase];
    int r = k / 5, s = k - r * 5;
    float ys = (float)(yy - 2 + r) + ofy;
    float xs = (float)(xx - 2 + s) + ofx;
    float y0f = floorf(ys), x0f = floorf(xs);
    float wy = ys - y0f, wx = xs - x0f;
    int y0 = (int)y0f, x0 = (int)x0f;
    int y1 = y0 + 1,  x1 = x0 + 1;
    float vy0 = (y0 >= 0 && y0 < Hh) ? 1.0f : 0.0f;
    float vy1 = (y1 >= 0 && y1 < Hh) ? 1.0f : 0.0f;
    float vx0 = (x0 >= 0 && x0 < Ww) ? 1.0f : 0.0f;
    float vx1 = (x1 >= 0 && x1 < Ww) ? 1.0f : 0.0f;
    float w00 = (1.0f - wy) * (1.0f - wx) * vy0 * vx0;
    float w01 = (1.0f - wy) * wx * vy0 * vx1;
    float w10 = wy * (1.0f - wx) * vy1 * vx0;
    float w11 = wy * wx * vy1 * vx1;
    int cy0 = imin(imax(y0, 0), Hh - 1);
    int cy1 = imin(imax(y1, 0), Hh - 1);
    int cx0 = imin(imax(x0, 0), Ww - 1);
    int cx1 = imin(imax(x1, 0), Ww - 1);
    int i00 = cy0 * Ww + cx0, i01 = cy0 * Ww + cx1;
    int i10 = cy1 * Ww + cx0, i11 = cy1 * Ww + cx1;
    const float* xg = x + (size_t)b * 64 * hw + (size_t)dg * 16 * hw;
    for (int cg = 0; cg < 16; ++cg) {
      const float* xb = xg + (size_t)cg * hw;
      float v = w00 * xb[i00] + w01 * xb[i01] + w10 * xb[i10] + w11 * xb[i11];
      samp[n][(dg * 16 + cg) * 25 + k] = (_Float16)(v * mm);
    }
  }
  __syncthreads();

  v8f acc = {};
  constexpr int NK = 50;
  for (int kc = 0; kc < NK; ++kc) {
    v16h afrag = *(const v16h*)(wpack + (((size_t)wv * NK + kc) * 32 + lane) * 16);
    const v16h* bp = (const v16h*)&samp[lane & 15][kc * 32 + ((lane >> 4) << 4)];
    acc = __builtin_amdgcn_wmma_f32_16x16x32_f16(false, afrag, false, *bp,
                                                 (short)0, acc, false, false);
  }
  int np = tile0 + (lane & 15);
  int b  = np / hw;
  int rp = np % hw;
  int yy = rp / Ww;
  int xx = rp % Ww;
#pragma unroll
  for (int v = 0; v < 8; ++v) {
    int M = wv * 16 + v + ((lane >> 4) << 3);
    out[(((size_t)b * 64 + M) * Hh + yy) * Ww + xx] = acc[v] + bias[M];
  }
}

// ---------------------------------------------------------------------------
// Bilinear 2x upsample (half-pixel centers, edge clamp), dims compile-time.
// ---------------------------------------------------------------------------
template <int HI, int WI>
__global__ __launch_bounds__(256) void up2_kernel(const float* __restrict__ in,
                                                  float* __restrict__ out) {
  constexpr int HO = HI * 2, WO = WI * 2;
  constexpr long total = (long)4 * 64 * HO * WO;
  long idx = (long)blockIdx.x * 256 + threadIdx.x;
  if (idx >= total) return;
  int ox = (int)(idx % WO);
  int oy = (int)((idx / WO) % HO);
  long bc = idx / ((long)WO * HO);
  float sy = (oy + 0.5f) * 0.5f - 0.5f;
  float sx = (ox + 0.5f) * 0.5f - 0.5f;
  float y0f = floorf(sy), x0f = floorf(sx);
  float wy = sy - y0f, wx = sx - x0f;
  int y0 = imin(imax((int)y0f, 0), HI - 1);
  int y1 = imin(imax((int)y0f + 1, 0), HI - 1);
  int x0 = imin(imax((int)x0f, 0), WI - 1);
  int x1 = imin(imax((int)x0f + 1, 0), WI - 1);
  const float* p = in + bc * HI * WI;
  float v = (1.0f - wy) * ((1.0f - wx) * p[y0 * WI + x0] + wx * p[y0 * WI + x1]) +
            wy * ((1.0f - wx) * p[y1 * WI + x0] + wx * p[y1 * WI + x1]);
  out[idx] = v;
}

// ---------------------------------------------------------------------------
// t-conditioned MLPs of tmb: mv = sigmoid(tp2 @ lrelu(tp1*t)),
// kern = lrelu(k2 @ lrelu(k1*t)).  One block per batch, 64 threads.
// ---------------------------------------------------------------------------
__global__ __launch_bounds__(64) void tmb_mlps_kernel(
    const float* __restrict__ tarr, int ti,
    const float* __restrict__ tp1, const float* __restrict__ tp2,
    const float* __restrict__ k1, const float* __restrict__ k2,
    float* __restrict__ mv, float* __restrict__ kern) {
  int b = blockIdx.x;
  int tid = threadIdx.x;
  __shared__ float h1[64];
  __shared__ float g1[64];
  float tv = tarr[b * 2 + ti];
  {
    float a = tp1[tid] * tv; h1[tid] = a >= 0.0f ? a : 0.1f * a;
    float c = k1[tid]  * tv; g1[tid] = c >= 0.0f ? c : 0.1f * c;
  }
  __syncthreads();
  float s = 0.0f;
  for (int c = 0; c < 64; ++c) s += tp2[tid * 64 + c] * h1[c];
  mv[b * 64 + tid] = 1.0f / (1.0f + __expf(-s));
  for (int j = 0; j < 9; ++j) {
    int o = j * 64 + tid;
    float s2 = 0.0f;
    for (int c = 0; c < 64; ++c) s2 += k2[o * 64 + c] * g1[c];
    kern[b * 576 + o] = s2 >= 0.0f ? s2 : 0.1f * s2;
  }
}

// ---------------------------------------------------------------------------
// tmb tail: per-(b,c) dynamic depthwise 3x3 on feat, * mv, + c33 conv result.
// ---------------------------------------------------------------------------
__global__ __launch_bounds__(256) void tmb_finish_kernel(
    const float* __restrict__ feat, const float* __restrict__ kern,
    const float* __restrict__ mv, const float* __restrict__ c33o,
    float* __restrict__ out) {
  constexpr int Hh = 64, Ww = 64;
  constexpr long total = (long)4 * 64 * Hh * Ww;
  long idx = (long)blockIdx.x * 256 + threadIdx.x;
  if (idx >= total) return;
  int xx = (int)(idx % Ww);
  int yy = (int)((idx / Ww) % Hh);
  int c  = (int)((idx / (Ww * Hh)) % 64);
  int b  = (int)(idx / ((long)Ww * Hh * 64));
  const float* kb = kern + b * 576 + c * 9;
  const float* fb = feat + ((size_t)b * 64 + c) * Hh * Ww;
  float acc = 0.0f;
#pragma unroll
  for (int r = 0; r < 3; ++r)
#pragma unroll
    for (int s = 0; s < 3; ++s) {
      int iy = yy + r - 1, ix = xx + s - 1;
      if (iy >= 0 && iy < Hh && ix >= 0 && ix < Ww)
        acc += fb[iy * Ww + ix] * kb[r * 3 + s];
    }
  out[idx] = acc * mv[b * 64 + c] + c33o[idx];
}

// ---------------------------------------------------------------------------
// Host orchestration
// ---------------------------------------------------------------------------
extern "C" void kernel_launch(void* const* d_in, const int* in_sizes, int n_in,
                              void* d_out, int out_size, void* d_ws, size_t ws_size,
                              hipStream_t stream) {
  (void)in_sizes; (void)n_in; (void)out_size; (void)ws_size;
  const int B_ = 4, NF_ = 64, H_ = 64, W_ = 64, TN_ = 2;

  const float* fea1 = (const float*)d_in[0];
  const float* fea2 = (const float*)d_in[1];
  const float* tA   = (const float*)d_in[2];   // [B,TN]
  const float* tBk  = (const float*)d_in[3];
  auto P = [&](int i) { return (const float*)d_in[i]; };

  size_t off = 0;
  auto alloc = [&](size_t bytes) -> char* {
    char* p = (char*)d_ws + off;
    off += (bytes + 255) & ~(size_t)255;
    return p;
  };

  const size_t S1 = (size_t)B_ * NF_ * H_ * W_;       // 1,048,576
  const size_t S2 = (size_t)B_ * NF_ * 32 * 32;
  const size_t S3 = (size_t)B_ * NF_ * 16 * 16;
  const size_t SH = (size_t)B_ * 100 * H_ * W_;

  float* base1 = (float*)alloc(S1 * 4);
  float* base2 = (float*)alloc(S1 * 4);
  float* bufA  = (float*)alloc(S1 * 4);
  float* bufB  = (float*)alloc(S1 * 4);
  float* bufC  = (float*)alloc(S1 * 4);
  float* otbuf = (float*)alloc(S1 * 4);
  float* o1buf = (float*)alloc(S1 * 4);
  float* a1buf = (float*)alloc(S1 * 4);
  float* a2buf = (float*)alloc(S1 * 4);
  float* h1 = (float*)alloc(S2 * 4);
  float* h2 = (float*)alloc(S2 * 4);
  float* h3 = (float*)alloc(S2 * 4);
  float* q1 = (float*)alloc(S3 * 4);
  float* q2 = (float*)alloc(S3 * 4);
  float* mbuf  = (float*)alloc(SH * 4);
  float* oxbuf = (float*)alloc(SH * 4);
  float* oybuf = (float*)alloc(SH * 4);
  float* mv   = (float*)alloc((size_t)B_ * 64 * 4);
  float* kern = (float*)alloc((size_t)B_ * 576 * 4);

  // ---- pack all conv weights to f16 WMMA fragments ----
  struct PackDef { int idx, Cout, Cin, ks; };
  const PackDef pd[23] = {
      {4, 64, 128, 3},  {6, 64, 64, 3},   {8, 64, 64, 3},  {10, 64, 64, 3},
      {12, 64, 64, 3},  {14, 64, 64, 3},  {16, 64, 64, 3}, {18, 64, 64, 3},
      {20, 64, 64, 3},  {22, 64, 64, 3},  {24, 64, 128, 3},{26, 64, 128, 3},
      {30, 64, 64, 3},  {32, 64, 64, 3},  {36, 64, 64, 3},
      {38, 64, 64, 3},  {40, 100, 64, 3}, {42, 64, 64, 3}, {44, 100, 64, 3},
      {46, 64, 64, 3},  {48, 100, 64, 3}, {50, 64, 64, 5}, {52, 64, 128, 1}};
  _Float16* pw[23];
  for (int i = 0; i < 23; ++i) {
    int Kin = pd[i].Cin * pd[i].ks * pd[i].ks;
    int nK = (Kin + 31) / 32, nM = (pd[i].Cout + 15) / 16;
    size_t tot = (size_t)nM * nK * 512;
    pw[i] = (_Float16*)alloc(tot * sizeof(_Float16));
    int blocks = (int)((tot + 255) / 256);
    if (blocks > 1024) blocks = 1024;
    pack_w_kernel<<<blocks, 256, 0, stream>>>(P(pd[i].idx), pw[i], pd[i].Cout, Kin);
  }

  auto up2a = [&](const float* in, float* out) {   // 16 -> 32
    constexpr long total = (long)4 * 64 * 32 * 32;
    up2_kernel<16, 16><<<(int)((total + 255) / 256), 256, 0, stream>>>(in, out);
  };
  auto up2b = [&](const float* in, float* out) {   // 32 -> 64
    constexpr long total = (long)4 * 64 * 64 * 64;
    up2_kernel<32, 32><<<(int)((total + 255) / 256), 256, 0, stream>>>(in, out);
  };

  // ---- pyramid(fa, fb) -> base ----
  auto pyramid = [&](const float* fa, const float* fb, float* base) {
    launch_conv<3,1,64,128,64,64,64>(fa, fb,    pw[0],  P(5),  bufA,  1, stream); // oc1_1
    launch_conv<3,1,64, 64,64,64,64>(bufA, bufA, pw[1],  P(7),  o1buf, 1, stream); // oc2_1 -> o1
    launch_conv<3,2,64, 64,64,64,64>(o1buf,o1buf,pw[2],  P(9),  h1,    1, stream); // oc3_1 s2
    launch_conv<3,1,64, 64,64,32,32>(h1, h1,    pw[3],  P(11), h2,    1, stream); // oc4_1
    launch_conv<3,1,64, 64,64,32,32>(h2, h2,    pw[4],  P(13), h1,    1, stream); // oc5_1 -> o2
    launch_conv<3,2,64, 64,64,32,32>(h1, h1,    pw[5],  P(15), q1,    1, stream); // oc6_1 s2
    launch_conv<3,1,64, 64,64,16,16>(q1, q1,    pw[6],  P(17), q2,    1, stream); // oc7_1
    launch_conv<3,1,64, 64,64,16,16>(q2, q2,    pw[7],  P(19), q1,    1, stream); // oc8_1 -> o3
    up2a(q1, h2);                                                                 // u (h2)
    launch_conv<3,1,64,128,64,32,32>(h2, h1,    pw[10], P(25), h3,    1, stream); // oc1_2 cat(u,o2)
    launch_conv<3,1,64, 64,64,32,32>(h3, h3,    pw[8],  P(21), h2,    1, stream); // oc2_2
    up2b(h2, bufA);                                                               // u (bufA)
    launch_conv<3,1,64,128,64,64,64>(bufA, o1buf, pw[11], P(27), bufB, 1, stream);// oc3_2 cat(u,o1)
    launch_conv<3,1,64, 64,64,64,64>(bufB, bufB, pw[9],  P(23), base,  0, stream);// oc4_2 (no act)
  };

  pyramid(fea1, fea2, base1);
  pyramid(fea2, fea1, base2);

  // ---- per-direction tmb + heads + dcn ----
  auto branch = [&](const float* base, const float* fea_src, const float* tarr,
                    int ti, float* aout) {
    launch_conv<3,1,64,64,64,64,64>(base, base, pw[12], P(31), bufA, 1, stream);  // fp1
    launch_conv<3,1,64,64,64,64,64>(bufA, bufA, pw[13], P(33), bufB, 1, stream);  // fp2 -> feat
    launch_conv<3,1,64,64,64,64,64>(base, base, pw[14], P(37), bufC, 0, stream);  // c33
    tmb_mlps_kernel<<<B_, 64, 0, stream>>>(tarr, ti, P(28), P(29), P(34), P(35),
                                           mv, kern);
    tmb_finish_kernel<<<(int)((S1 + 255) / 256), 256, 0, stream>>>(bufB, kern, mv,
                                                                   bufC, otbuf);
    // heads (mask -> sigmoid, ox/oy -> linear)
    launch_conv<3,1,64,64, 64,64,64>(otbuf, otbuf, pw[15], P(39), bufA,  1, stream);
    launch_conv<3,1,64,64,100,64,64>(bufA, bufA,   pw[16], P(41), mbuf,  2, stream);
    launch_conv<3,1,64,64, 64,64,64>(otbuf, otbuf, pw[17], P(43), bufA,  1, stream);
    launch_conv<3,1,64,64,100,64,64>(bufA, bufA,   pw[18], P(45), oxbuf, 0, stream);
    launch_conv<3,1,64,64, 64,64,64>(otbuf, otbuf, pw[19], P(47), bufA,  1, stream);
    launch_conv<3,1,64,64,100,64,64>(bufA, bufA,   pw[20], P(49), oybuf, 0, stream);
    // deformable conv: 1024 16-pixel tiles
    dcn_wmma_kernel<<<(B_ * H_ * W_) / 16, 128, 0, stream>>>(
        fea_src, oxbuf, oybuf, mbuf, pw[21], P(51), aout);
  };

  for (int i = 0; i < TN_; ++i) {
    branch(base1, fea1, tA, i, a2buf);   // a2 = dcn(fea1, ...)
    branch(base2, fea2, tBk, i, a1buf);  // a1 = dcn(fea2, ...)
    float* outp = (float*)d_out + (size_t)i * S1;
    launch_conv<1,1,64,128,64,64,64>(a1buf, a2buf, pw[22], P(53), outp, 0, stream); // fus
  }
}
